// ResConv3DAttnBlock_50251117363265
// MI455X (gfx1250) — matrix operand order repaired and checked by reading, not compiled, not run on hardware
//
#include <hip/hip_runtime.h>
#include <math.h>

typedef __bf16 bf16;
typedef __attribute__((ext_vector_type(16))) __bf16 v16bf;
typedef __attribute__((ext_vector_type(4)))  __bf16 v4bf;
typedef __attribute__((ext_vector_type(8)))  float  v8f;
typedef __attribute__((ext_vector_type(4)))  unsigned int v4u;
typedef __attribute__((ext_vector_type(8)))  int    v8i;
typedef __attribute__((ext_vector_type(4)))  int    v4i;

#define NB   2
#define CI   64
#define CB   128
#define CO   128
#define AH   32
#define DD   9
#define HH   160
#define WW   160
#define PP   (HH*WW)        // 25600
#define SPD  (DD*PP)        // 230400
#define EPSI 1e-5f

__device__ __forceinline__ float gelu_f(float v) {
  return 0.5f * v * (1.0f + erff(v * 0.70710678118654752f));
}

// ---- WMMA fragment loaders (CDNA5 16-bit 16x32 A / 32x16 B layouts, ISA 7.12.2) ----
__device__ __forceinline__ v16bf ld_afrag(const bf16* base, int ld, int row0, int k0, int lane) {
  int half = lane >> 4;
  const bf16* r = base + (row0 + (lane & 15)) * ld;
  v16bf a;
#pragma unroll
  for (int j = 0; j < 8; ++j) {
    int k = k0 + ((j < 4) ? 0 : 16) + 8 * half + 2 * (j & 3);
    a[2 * j]     = r[k];
    a[2 * j + 1] = r[k + 1];
  }
  return a;
}
__device__ __forceinline__ v16bf ld_bfrag(const bf16* base, int ld, int col0, int k0, int lane) {
  int half = lane >> 4;
  const bf16* r = base + (col0 + (lane & 15)) * ld;
  v16bf b;
#pragma unroll
  for (int j = 0; j < 8; ++j) {
    int k = k0 + ((j < 4) ? 0 : 16) + 8 * half + 2 * (j & 3);
    b[2 * j]     = r[k];
    b[2 * j + 1] = r[k + 1];
  }
  return b;
}

// ---------------- kernel 0: zero the stats scratch ----------------
__global__ void zero_stats(float* st) {
  int i = blockIdx.x * 256 + threadIdx.x;
  if (i < 3 * 2 * NB * CB) st[i] = 0.0f;
}

// ---------------- kernel 1: conv1 (1x1x1) as WMMA GEMM + instnorm stats ----------------
__global__ __launch_bounds__(256) void conv1_gemm(const float* __restrict__ x,
                                                  const float* __restrict__ w,
                                                  bf16* __restrict__ h1,
                                                  float* __restrict__ st) {
  __shared__ bf16 As[CB * 66];
  __shared__ bf16 Bs[64 * 66];
  __shared__ float ssum[CB], ssq[CB];
  int tid = threadIdx.x;
  int t0  = blockIdx.x * 64;
  int n   = t0 / SPD;
  int s0  = t0 % SPD;
  if (tid < CB) { ssum[tid] = 0.0f; ssq[tid] = 0.0f; }
  const float* xb = x + (size_t)n * CI * SPD + s0;
  __builtin_prefetch(xb, 0, 0);               // global_prefetch_b8
  for (int idx = tid; idx < CB * CI; idx += 256) {
    int m = idx >> 6, k = idx & 63;
    As[m * 66 + k] = (bf16)w[idx];
  }
  for (int idx = tid; idx < 64 * CI; idx += 256) {
    int ci = idx >> 6, j = idx & 63;
    Bs[j * 66 + ci] = (bf16)xb[(size_t)ci * SPD + j];
  }
  __syncthreads();

  int wid = tid >> 5, lane = tid & 31;
  int half = lane >> 4, lq = lane & 15;
  v8f acc[4] = {};
#pragma unroll
  for (int kk = 0; kk < 64; kk += 32) {
    v16bf a = ld_afrag(As, 66, wid * 16, kk, lane);
#pragma unroll
    for (int nt = 0; nt < 4; ++nt) {
      v16bf b = ld_bfrag(Bs, 66, nt * 16, kk, lane);
      acc[nt] = __builtin_amdgcn_wmma_f32_16x16x32_bf16(false, a, false, b,
                                                        (short)0, acc[nt], false, false);
    }
  }
  float ps[8] = {0, 0, 0, 0, 0, 0, 0, 0}, pq[8] = {0, 0, 0, 0, 0, 0, 0, 0};
#pragma unroll
  for (int nt = 0; nt < 4; ++nt) {
    int col = s0 + nt * 16 + lq;
#pragma unroll
    for (int j = 0; j < 8; ++j) {
      int m   = wid * 16 + j + 8 * half;
      bf16 vb = (bf16)acc[nt][j];
      h1[(size_t)(n * CB + m) * SPD + col] = vb;
      float vf = (float)vb;
      ps[j] += vf;
      pq[j] += vf * vf;
    }
  }
#pragma unroll
  for (int j = 0; j < 8; ++j) {
    float s = ps[j], q = pq[j];
#pragma unroll
    for (int msk = 8; msk >= 1; msk >>= 1) {
      s += __shfl_xor(s, msk, 32);
      q += __shfl_xor(q, msk, 32);
    }
    if (lq == 0) {
      int m = wid * 16 + j + 8 * half;
      atomicAdd(&ssum[m], s);
      atomicAdd(&ssq[m], q);
    }
  }
  __syncthreads();
  if (tid < CB) {
    atomicAdd(&st[n * CB + tid], ssum[tid]);
    atomicAdd(&st[NB * CB + n * CB + tid], ssq[tid]);
  }
}

// ---------------- instance-norm (+ optional exact GELU), in place ----------------
__global__ __launch_bounds__(256) void norm_gelu(bf16* __restrict__ buf,
                                                 const float* __restrict__ st,
                                                 const float* __restrict__ gamma,
                                                 const float* __restrict__ beta,
                                                 int do_gelu) {
  long idx = ((long)blockIdx.x * 256 + threadIdx.x) * 4;
  int nc = (int)(idx / SPD);
  int c  = nc & 127;
  float mean  = st[nc] * (1.0f / SPD);
  float var   = st[NB * CB + nc] * (1.0f / SPD) - mean * mean;
  float rstd  = rsqrtf(var + EPSI);
  float scale = gamma[c] * rstd;
  float shift = beta[c] - mean * scale;
  v4bf v = *(const v4bf*)(buf + idx);
#pragma unroll
  for (int j = 0; j < 4; ++j) {
    float f = (float)v[j] * scale + shift;
    if (do_gelu) f = gelu_f(f);
    v[j] = (bf16)f;
  }
  *(v4bf*)(buf + idx) = v;
}

// ---------------- grouped (1,3,3) conv, replicate pad, + stats ----------------
__global__ __launch_bounds__(160) void conv2_grouped(const bf16* __restrict__ h1,
                                                     const float* __restrict__ w2,
                                                     bf16* __restrict__ h2,
                                                     float* __restrict__ st) {
  __shared__ bf16  sin[4][3][164];
  __shared__ float swt[4][4][9];
  __shared__ float ssum[4], ssq[4];
  int tid = threadIdx.x;
  int bid = blockIdx.x;
  int h = bid % HH; int t = bid / HH;
  int d = t % DD;   t /= DD;
  int g = t % 32;   int n = t / 32;
  if (tid < 4) { ssum[tid] = 0.0f; ssq[tid] = 0.0f; }
  if (tid < 144) {
    int co = tid / 36, r = tid % 36;
    swt[co][r / 9][r % 9] = w2[(g * 4 + co) * 36 + r];
  }
  for (int idx = tid; idx < 4 * 3 * 162; idx += 160) {
    int ci = idx / 486, rr = (idx / 162) % 3, j = idx % 162;
    int hh2 = min(max(h - 1 + rr, 0), HH - 1);
    int ww2 = min(max(j - 1, 0), WW - 1);
    sin[ci][rr][j] = h1[(size_t)(n * CB + g * 4 + ci) * SPD + d * PP + hh2 * WW + ww2];
  }
  __syncthreads();
  int wq = tid;
  float o[4] = {0, 0, 0, 0};
#pragma unroll
  for (int ci = 0; ci < 4; ++ci)
#pragma unroll
    for (int kh = 0; kh < 3; ++kh)
#pragma unroll
      for (int kw = 0; kw < 3; ++kw) {
        float iv = (float)sin[ci][kh][wq + kw];
#pragma unroll
        for (int co = 0; co < 4; ++co) o[co] += iv * swt[co][ci][kh * 3 + kw];
      }
#pragma unroll
  for (int co = 0; co < 4; ++co) {
    bf16 vb = (bf16)o[co];
    h2[(size_t)(n * CB + g * 4 + co) * SPD + d * PP + h * WW + wq] = vb;
    float vf = (float)vb, s = vf, q = vf * vf;
#pragma unroll
    for (int msk = 16; msk >= 1; msk >>= 1) {
      s += __shfl_xor(s, msk, 32);
      q += __shfl_xor(q, msk, 32);
    }
    if ((tid & 31) == 0) { atomicAdd(&ssum[co], s); atomicAdd(&ssq[co], q); }
  }
  __syncthreads();
  if (tid < 4) {
    int c = g * 4 + tid;
    atomicAdd(&st[n * CB + c], ssum[tid]);
    atomicAdd(&st[NB * CB + n * CB + c], ssq[tid]);
  }
}

// ---------------- fused Q/K/V projections (M=192) WMMA GEMM ----------------
__global__ __launch_bounds__(256) void qkv_gemm(const bf16* __restrict__ h2,
                                                const float* __restrict__ wq,
                                                const float* __restrict__ wk,
                                                const float* __restrict__ wvv,
                                                const float* __restrict__ qe,
                                                const float* __restrict__ ke,
                                                bf16* __restrict__ Qb, bf16* __restrict__ Kb,
                                                bf16* __restrict__ Vb, float* __restrict__ st) {
  __shared__ bf16 As[192 * 130];
  __shared__ bf16 Bs[32 * 130];
  __shared__ float ssum[CB], ssq[CB];
  int tid = threadIdx.x;
  int t0  = blockIdx.x * 32;
  int n   = t0 / SPD;
  int s0  = t0 % SPD;
  int d   = s0 / PP;
  if (tid < CB) { ssum[tid] = 0.0f; ssq[tid] = 0.0f; }
  for (int idx = tid; idx < 192 * 128; idx += 256) {
    int m = idx >> 7;
    float v;
    if (m < 32)      v = wq[idx];
    else if (m < 64) v = wk[idx - 32 * 128];
    else             v = wvv[idx - 64 * 128];
    As[m * 130 + (idx & 127)] = (bf16)v;
  }
  const bf16* hb = h2 + (size_t)n * CB * SPD + s0;
  __builtin_prefetch(hb, 0, 0);
  for (int idx = tid; idx < 32 * 128; idx += 256) {
    int k = idx >> 5, col = idx & 31;
    Bs[col * 130 + k] = hb[(size_t)k * SPD + col];
  }
  __syncthreads();

  int wid = tid >> 5, lane = tid & 31;
  int half = lane >> 4, lq = lane & 15;
  int mw = wid >> 1, nw = wid & 1;
  v8f acc[3] = {};
#pragma unroll
  for (int kk = 0; kk < 128; kk += 32) {
    v16bf b = ld_bfrag(Bs, 130, nw * 16, kk, lane);
#pragma unroll
    for (int i = 0; i < 3; ++i) {
      v16bf a = ld_afrag(As, 130, (mw * 3 + i) * 16, kk, lane);
      acc[i]  = __builtin_amdgcn_wmma_f32_16x16x32_bf16(false, a, false, b,
                                                        (short)0, acc[i], false, false);
    }
  }
#pragma unroll
  for (int i = 0; i < 3; ++i) {
    int m0  = (mw * 3 + i) * 16;
    int col = s0 + nw * 16 + lq;
    float ps[8] = {0, 0, 0, 0, 0, 0, 0, 0}, pq[8] = {0, 0, 0, 0, 0, 0, 0, 0};
#pragma unroll
    for (int j = 0; j < 8; ++j) {
      int m   = m0 + j + 8 * half;
      float v = acc[i][j];
      if (m0 < 32) {
        v += qe[m * DD + d];
        Qb[(size_t)(n * AH + m) * SPD + col] = (bf16)v;
      } else if (m0 < 64) {
        v += ke[(m - 32) * DD + d];
        Kb[(size_t)(n * AH + (m - 32)) * SPD + col] = (bf16)v;
      } else {
        bf16 vb = (bf16)v;
        Vb[(size_t)(n * CB + (m - 64)) * SPD + col] = vb;
        float vf = (float)vb;
        ps[j] = vf;
        pq[j] = vf * vf;
      }
    }
    if (m0 >= 64) {
#pragma unroll
      for (int j = 0; j < 8; ++j) {
        float s = ps[j], q = pq[j];
#pragma unroll
        for (int msk = 8; msk >= 1; msk >>= 1) {
          s += __shfl_xor(s, msk, 32);
          q += __shfl_xor(q, msk, 32);
        }
        if (lq == 0) {
          int c = m0 - 64 + j + 8 * half;
          atomicAdd(&ssum[c], s);
          atomicAdd(&ssq[c], q);
        }
      }
    }
  }
  __syncthreads();
  if (tid < CB) {
    atomicAdd(&st[n * CB + tid], ssum[tid]);
    atomicAdd(&st[NB * CB + n * CB + tid], ssq[tid]);
  }
}

// ---------------- per-pixel 9x9 scores over A=32 + softmax ----------------
__global__ __launch_bounds__(256) void attn_scores(const bf16* __restrict__ Qb,
                                                   const bf16* __restrict__ Kb,
                                                   float* __restrict__ attn) {
  int g  = blockIdx.x * 256 + threadIdx.x;
  int n  = g / PP, hw = g % PP;
  float sc[81];
#pragma unroll
  for (int i = 0; i < 81; ++i) sc[i] = 0.0f;
  const bf16* qp = Qb + (size_t)n * AH * SPD + hw;
  const bf16* kp = Kb + (size_t)n * AH * SPD + hw;
  for (int a = 0; a < AH; ++a) {
    float qa[9], ka[9];
#pragma unroll
    for (int d2 = 0; d2 < 9; ++d2) {
      qa[d2] = (float)qp[(size_t)a * SPD + d2 * PP];
      ka[d2] = (float)kp[(size_t)a * SPD + d2 * PP];
    }
#pragma unroll
    for (int d2 = 0; d2 < 9; ++d2)
#pragma unroll
      for (int e = 0; e < 9; ++e) sc[d2 * 9 + e] += qa[d2] * ka[e];
  }
  const float inv = 0.17677669529663687f;
#pragma unroll
  for (int d2 = 0; d2 < 9; ++d2) {
    float mx = -1e30f;
#pragma unroll
    for (int e = 0; e < 9; ++e) mx = fmaxf(mx, sc[d2 * 9 + e] * inv);
    float sum = 0.0f;
#pragma unroll
    for (int e = 0; e < 9; ++e) {
      float ex = __expf(sc[d2 * 9 + e] * inv - mx);
      sc[d2 * 9 + e] = ex;
      sum += ex;
    }
    float r = 1.0f / sum;
#pragma unroll
    for (int e = 0; e < 9; ++e)
      attn[(size_t)(n * 81 + d2 * 9 + e) * PP + hw] = sc[d2 * 9 + e] * r;
  }
}

// ---------------- out = gelu(x_pad + attn @ V); attn tile fetched by TDM ----------------
__global__ __launch_bounds__(256) void attn_apply(const float* __restrict__ attn,
                                                  const bf16* __restrict__ Vb,
                                                  const float* __restrict__ x,
                                                  float* __restrict__ out) {
  __shared__ __align__(16) float At[81][32];
  int tid = threadIdx.x;
  int bid = blockIdx.x;
  int n   = bid / (PP / 32);
  int hw0 = (bid % (PP / 32)) * 32;

  // Tensor Data Mover: 2D tile (81 rows x 32 cols, f32) from strided global into
  // contiguous LDS.  D# per CDNA5 ISA; wave 0 issues, EXEC
  // ignored by TDM; completion via TENSORcnt, then block barrier publishes LDS.
  if (tid < 32) {
    unsigned lds_off = (unsigned)(size_t)(void*)&At[0][0];
    unsigned long long ga =
        (unsigned long long)(size_t)(const void*)(attn + (size_t)n * 81 * PP + hw0);
    // group0: count=1 | lds_addr | global_addr[56:0] | type=2
    v4u g0 = { 1u, lds_off, (unsigned)ga,
               (unsigned)(((ga >> 32) & 0x01FFFFFFull)) | 0x80000000u };
    // group1: data_size=2 (4B); tensor_dim0=PP; tensor_dim1=81; tile_dim0=32;
    //         tile_dim1=81; tensor_dim0_stride=PP
    v8i g1 = { (int)(2u << 16),
               (int)(((unsigned)PP & 0xFFFFu) << 16),
               (int)(((unsigned)PP >> 16) | (81u << 16)),
               (int)(32u << 16),
               81,
               PP,
               0, 0 };
    v4i gz = { 0, 0, 0, 0 };
#if __clang_major__ >= 23
    v8i g4 = { 0, 0, 0, 0, 0, 0, 0, 0 };
    __builtin_amdgcn_tensor_load_to_lds(g0, g1, gz, gz, g4, 0);
#else
    __builtin_amdgcn_tensor_load_to_lds(g0, g1, gz, gz, 0);
#endif
    __builtin_amdgcn_s_wait_tensorcnt(0);
  }
  __syncthreads();

  int hl = tid & 31;
  int c0 = (tid >> 5) * 16;
  int hw = hw0 + hl;
  for (int cc = 0; cc < 16; ++cc) {
    int c = c0 + cc;
    float v[9];
#pragma unroll
    for (int e = 0; e < 9; ++e)
      v[e] = (float)Vb[(size_t)(n * CB + c) * SPD + e * PP + hw];
#pragma unroll
    for (int d2 = 0; d2 < 9; ++d2) {
      float acc = 0.0f;
#pragma unroll
      for (int e = 0; e < 9; ++e) acc += At[d2 * 9 + e][hl] * v[e];
      float xi = (c < CI) ? x[(size_t)(n * CI + c) * SPD + d2 * PP + hw] : 0.0f;
      out[(size_t)(n * CO + c) * SPD + d2 * PP + hw] = gelu_f(xi + acc);
    }
  }
}

extern "C" void kernel_launch(void* const* d_in, const int* in_sizes, int n_in,
                              void* d_out, int out_size, void* d_ws, size_t ws_size,
                              hipStream_t stream) {
  (void)in_sizes; (void)n_in; (void)out_size; (void)ws_size;
  const float* x  = (const float*)d_in[0];
  const float* w1 = (const float*)d_in[1];
  const float* g1 = (const float*)d_in[2];
  const float* b1 = (const float*)d_in[3];
  const float* w2 = (const float*)d_in[4];
  const float* g2 = (const float*)d_in[5];
  const float* b2 = (const float*)d_in[6];
  const float* wq = (const float*)d_in[7];
  const float* wk = (const float*)d_in[8];
  const float* wv = (const float*)d_in[9];
  const float* gv = (const float*)d_in[10];
  const float* bv = (const float*)d_in[11];
  const float* qe = (const float*)d_in[12];
  const float* ke = (const float*)d_in[13];
  float* out = (float*)d_out;

  float* st = (float*)d_ws;
  char* base = (char*)d_ws + 8192;
  size_t HS = (size_t)NB * CB * SPD;
  size_t QS = (size_t)NB * AH * SPD;
  bf16* h1 = (bf16*)base;
  bf16* h2 = h1 + HS;
  bf16* Vb = h2 + HS;
  bf16* Qb = Vb + HS;
  bf16* Kb = Qb + QS;
  float* attn = (float*)(Kb + QS);

  zero_stats<<<6, 256, 0, stream>>>(st);
  conv1_gemm<<<NB * SPD / 64, 256, 0, stream>>>(x, w1, h1, st);
  norm_gelu<<<NB * CB * SPD / 1024, 256, 0, stream>>>(h1, st, g1, b1, 1);
  conv2_grouped<<<NB * 32 * DD * HH, 160, 0, stream>>>(h1, w2, h2, st + 2 * NB * CB);
  norm_gelu<<<NB * CB * SPD / 1024, 256, 0, stream>>>(h2, st + 2 * NB * CB, g2, b2, 1);
  qkv_gemm<<<NB * SPD / 32, 256, 0, stream>>>(h2, wq, wk, wv, qe, ke, Qb, Kb, Vb,
                                              st + 4 * NB * CB);
  norm_gelu<<<NB * CB * SPD / 1024, 256, 0, stream>>>(Vb, st + 4 * NB * CB, gv, bv, 0);
  attn_scores<<<NB * PP / 256, 256, 0, stream>>>(Qb, Kb, attn);
  attn_apply<<<NB * PP / 32, 256, 0, stream>>>(attn, Vb, x, out);
}
